// top_Block_58119497449919
// MI455X (gfx1250) — compile-verified, matrix operand order checked by reading, not run on hardware
//
#include <hip/hip_runtime.h>
#include <hip/hip_bf16.h>

// MI455X (gfx1250) BiFormer-style top_Block. All dense matmuls on
// v_wmma_f32_16x16x32_f16 (wave32, fp32 accum). Compute-bound (~49 GFLOP vs
// ~67MB HBM); intermediates live in L2. WMMA operands pre-packed into the ISA
// fragment layout (one 32B vector load per lane per fragment). Big GEMMs are
// M-blocked x2 with K-outer loops: each B fragment feeds two WMMAs and only
// the accumulators stay register-resident (no spill-prone A hoisting).

typedef _Float16 f16;
typedef __attribute__((ext_vector_type(16))) _Float16 v16h;
typedef __attribute__((ext_vector_type(8)))  _Float16 v8h;
typedef __attribute__((ext_vector_type(8)))  float    v8f;

#define DIM   256
#define QK    256
#define HEADS 8
#define TOPK  4
#define HID   1024
#define NB    8
#define P2    256
#define W2    16
#define HW    4096
#define NPIX  (NB*HW)
#define SCALE 0.0625f

// K-pattern of the 16-bit A/B fragment (ISA 7.12.2): element e of lane with
// half h = lane>>4 holds K = ((e<8)?e:e+8) + 8h.
__device__ __forceinline__ int kpat(int e, int h) {
  return ((e < 8) ? e : e + 8) + h * 8;
}
// Packed-tile index of (row 0..15, col 0..K-1): tile stored as
// [kk=col/32][lane=32][e=16] halves; inverse of kpat.
__device__ __forceinline__ int pack_idx(int row, int col) {
  int kk = col >> 5, cc = col & 31;
  int hl = (cc >> 3) & 1;
  int e  = (cc & 7) + ((cc >= 16) ? 8 : 0);
  return kk * 512 + (row + 16 * hl) * 16 + e;
}
__device__ __forceinline__ v16h frag_ld16(const f16* p) { return *(const v16h*)p; }
__device__ __forceinline__ v8f wmma16(v16h a, v16h b, v8f c) {
  return __builtin_amdgcn_wmma_f32_16x16x32_f16(false, a, false, b, (short)0, c,
                                                false, false);
}

// ---- 0) pack fp32 matrix [nout][K] into f16 fragment tiles ---------------
__global__ void pack_b_kernel(const float* __restrict__ in, f16* __restrict__ out,
                              int K, long total) {
  long idx = (long)blockIdx.x * 256 + threadIdx.x;
  if (idx >= total) return;
  int row = (int)(idx / K), col = (int)(idx % K);
  out[(size_t)(row >> 4) * K * 16 + pack_idx(row & 15, col)] = (f16)in[idx];
}

// ---- 1) window QKV projection + BN + window means -------------------------
// grid NB*P2/2 (2 windows per block), 256 thr (8 waves); 48 n-tiles, K=256.
// K-outer: per K-step load 2 A frags + 6 B frags -> 12 WMMAs.
__global__ void __launch_bounds__(256, 1)
proj_kernel(const float* __restrict__ x,
            const f16* __restrict__ wqp, const f16* __restrict__ wkvp,
            const float* __restrict__ gq, const float* __restrict__ bq,
            const float* __restrict__ gkv, const float* __restrict__ bkv,
            f16* __restrict__ qp, f16* __restrict__ kp,
            f16* __restrict__ vp, f16* __restrict__ v_h,
            float* __restrict__ qwin, float* __restrict__ kwin) {
  __shared__ __align__(32) f16 a_lds[2][W2 * DIM];  // 2 packed window tiles, 16KB
  int blk = blockIdx.x, n = blk >> 7, win0 = (blk & 127) * 2;
  int tid = threadIdx.x;
#pragma unroll
  for (int t = 0; t < 32; ++t) {  // window mw=t>>4, token t&15, channel tid
    int mw = t >> 4, tok = t & 15, win = win0 + mw;
    int y = (win >> 4) * 4 + (tok >> 2), xx = (win & 15) * 4 + (tok & 3);
    a_lds[mw][pack_idx(tok, tid)] = (f16)x[(((size_t)n * DIM + tid) * 64 + y) * 64 + xx];
  }
  __syncthreads();

  int wave = tid >> 5, lane = tid & 31, r = lane & 15, hlf = lane >> 4;
  const f16* bp[6];
#pragma unroll
  for (int tt = 0; tt < 6; ++tt) {
    int tile = wave + tt * 8;
    bp[tt] = ((tile < 16) ? (wqp + (size_t)tile * 4096)
                          : (wkvp + (size_t)(tile - 16) * 4096)) + lane * 16;
  }
  __builtin_prefetch(bp[0], 0, 0);

  v8f acc[6][2] = {};
#pragma unroll 2
  for (int kk = 0; kk < 8; ++kk) {
    v16h a0 = frag_ld16(&a_lds[0][kk * 512] + lane * 16);
    v16h a1 = frag_ld16(&a_lds[1][kk * 512] + lane * 16);
#pragma unroll
    for (int tt = 0; tt < 6; ++tt) {
      v16h b = frag_ld16(bp[tt] + kk * 512);
      acc[tt][0] = wmma16(a0, b, acc[tt][0]);
      acc[tt][1] = wmma16(a1, b, acc[tt][1]);
    }
  }

#pragma unroll
  for (int tt = 0; tt < 6; ++tt) {
    int tile = wave + tt * 8;  // [0,16)=q, [16,32)=k, [32,48)=v
    const float *g, *bb; int n0;
    if (tile < 16) { g = gq; bb = bq; n0 = tile * 16; }
    else           { g = gkv; bb = bkv; n0 = (tile - 16) * 16; }
    float gg = g[n0 + r], be = bb[n0 + r];
    int c = ((tile < 32) ? n0 : (n0 - QK)) + r;  // output channel 0..255
#pragma unroll
    for (int mw = 0; mw < 2; ++mw) {
      size_t wbase = (size_t)(n * P2 + win0 + mw);
      union { v8f v; float f[8]; } ua; ua.v = acc[tt][mw];
      if (tile < 32) {  // q/k -> packed per-(window,head) fragments
        f16* dst = (tile < 16) ? qp : kp;
        int head = c >> 5, cc = c & 31;
        int hl2 = (cc >> 3) & 1;
        int e2 = (cc & 7) + ((cc >= 16) ? 8 : 0);
        size_t pb = wbase * 4096 + head * 512 + e2;
        float colsum = 0.f;
#pragma unroll
        for (int i2 = 0; i2 < 8; ++i2) {
          int m = i2 + 8 * hlf;
          float val = ua.f[i2] * gg + be;
          dst[pb + (size_t)(m + 16 * hl2) * 16] = (f16)val;
          colsum += val;
        }
        colsum += __shfl_xor(colsum, 16);
        if (lane < 16) {
          float* wb = (tile < 16) ? qwin : kwin;
          wb[wbase * QK + c] = colsum * (1.f / 16.f);
        }
      } else {  // v -> packed B fragments [win][nn][head][lane][8] + linear copy
        int head = c >> 5, c5 = c & 31;
        int nn2 = c5 >> 4, r2 = c5 & 15;
        size_t pb = wbase * 4096 + (size_t)(nn2 * 8 + head) * 256;
#pragma unroll
        for (int i2 = 0; i2 < 8; ++i2) {
          int m = i2 + 8 * hlf;
          f16 hv = (f16)(ua.f[i2] * gg + be);
          vp[pb + (size_t)(r2 + 16 * (m >> 3)) * 8 + (m & 7)] = hv;
          v_h[(wbase * W2 + m) * DIM + c] = hv;
        }
      }
    }
  }
}

// ---- 2) routing logits: (qwin*SCALE) @ kwin^T -----------------------------
__global__ void route_kernel(const float* __restrict__ qwin,
                             const f16* __restrict__ kwp,
                             float* __restrict__ logits) {
  __shared__ __align__(32) f16 a_lds[16 * QK];
  int n = blockIdx.x >> 4, mt = blockIdx.x & 15;
  int tid = threadIdx.x;
  const float* qb = qwin + ((size_t)n * P2 + mt * 16) * QK;
#pragma unroll
  for (int t = 0; t < 16; ++t)
    a_lds[pack_idx(t, tid)] = (f16)(qb[(size_t)t * QK + tid] * SCALE);
  __syncthreads();

  int wave = tid >> 5, lane = tid & 31, r = lane & 15, hlf = lane >> 4;
  const f16* afrag = a_lds + lane * 16;
  for (int tt = 0; tt < 2; ++tt) {
    int nt = wave * 2 + tt;
    const f16* bpp = kwp + (size_t)(n * 16 + nt) * 4096 + lane * 16;
    v8f acc = {};
#pragma unroll
    for (int kk = 0; kk < 8; ++kk)
      acc = wmma16(frag_ld16(afrag + kk * 512), frag_ld16(bpp + kk * 512), acc);
    union { v8f v; float f[8]; } ua; ua.v = acc;
#pragma unroll
    for (int i2 = 0; i2 < 8; ++i2)
      logits[((size_t)n * P2 + mt * 16 + i2 + 8 * hlf) * P2 + nt * 16 + r] = ua.f[i2];
  }
}

// ---- 3) top-4 (ties -> lower index, matches lax.top_k) --------------------
__global__ void topk_kernel(const float* __restrict__ logits, int* __restrict__ ridx) {
  int row = blockIdx.x * 256 + threadIdx.x;
  if (row >= NB * P2) return;
  const float* l = logits + (size_t)row * P2;
  int cs[TOPK];
  for (int t = 0; t < TOPK; ++t) {
    float best = -3.4e38f; int bi = 0;
    for (int p = 0; p < P2; ++p) {
      bool skip = false;
      for (int u = 0; u < t; ++u) skip |= (cs[u] == p);
      float v = l[p];
      if (!skip && v > best) { best = v; bi = p; }
    }
    cs[t] = bi;
    ridx[row * TOPK + t] = bi;
  }
}

// ---- 4) attention: one wave per (window, head) ----------------------------
__global__ void attn_kernel(const f16* __restrict__ qp, const f16* __restrict__ kp,
                            const f16* __restrict__ vp, const int* __restrict__ ridx,
                            float* __restrict__ obuf) {
  __shared__ float s_lds[HEADS * 16 * 64];              // 32KB scores
  __shared__ __align__(32) f16 p_lds[HEADS * 16 * 64];  // 16KB packed probs
  int blk = blockIdx.x, n = blk >> 8, win = blk & 255;
  int tid = threadIdx.x, head = tid >> 5, lane = tid & 31;
  int r = lane & 15, hlf = lane >> 4;
  const int* idx = ridx + (size_t)(n * P2 + win) * TOPK;
  int sel[TOPK] = {idx[0], idx[1], idx[2], idx[3]};

  v16h aq = frag_ld16(qp + ((size_t)(n * P2 + win)) * 4096 + head * 512 + lane * 16);
  float* S = s_lds + head * 16 * 64;
  f16*   P = p_lds + head * 1024;
#pragma unroll
  for (int nt = 0; nt < TOPK; ++nt) {
    v16h bk = frag_ld16(kp + ((size_t)(n * P2 + sel[nt])) * 4096 + head * 512 + lane * 16);
    v8f acc = {};
    acc = wmma16(aq, bk, acc);
    union { v8f v; float f[8]; } ua; ua.v = acc;
#pragma unroll
    for (int i2 = 0; i2 < 8; ++i2) S[(i2 + 8 * hlf) * 64 + nt * 16 + r] = ua.f[i2];
  }
  __syncthreads();

  if (lane < 16) {  // softmax over 64 keys (attention scale applied here)
    float* row = S + lane * 64;
    float buf[64];
    float mx = -3.4e38f;
    for (int p = 0; p < 64; ++p) { buf[p] = row[p] * SCALE; mx = fmaxf(mx, buf[p]); }
    float s = 0.f;
    for (int p = 0; p < 64; ++p) { buf[p] = __expf(buf[p] - mx); s += buf[p]; }
    float inv = 1.f / s;
    for (int p = 0; p < 64; ++p) P[pack_idx(lane, p)] = (f16)(buf[p] * inv);
  }
  __syncthreads();

  int jj = win >> 4, ii = win & 15;
#pragma unroll
  for (int nn = 0; nn < 2; ++nn) {
    v8f acc = {};
#pragma unroll
    for (int kc = 0; kc < 2; ++kc) {
      v16h ap = frag_ld16(P + kc * 512 + lane * 16);
      // V frag: lo 8 halves from sel[2kc], hi 8 from sel[2kc+1]; lane'==lane.
      union { v16h v; v8h h[2]; } uv;
      size_t voff = (size_t)(nn * 8 + head) * 256 + lane * 8;
      uv.h[0] = *(const v8h*)(vp + ((size_t)(n * P2 + sel[2 * kc])) * 4096 + voff);
      uv.h[1] = *(const v8h*)(vp + ((size_t)(n * P2 + sel[2 * kc + 1])) * 4096 + voff);
      acc = wmma16(ap, uv.v, acc);
    }
    union { v8f v; float f[8]; } ua; ua.v = acc;
#pragma unroll
    for (int i2 = 0; i2 < 8; ++i2) {
      int m = i2 + 8 * hlf;
      int y = jj * 4 + (m >> 2), xx = ii * 4 + (m & 3);
      obuf[(((size_t)n * 64 + y) * 64 + xx) * DIM + head * 32 + nn * 16 + r] = ua.f[i2];
    }
  }
}

// ---- 5) LEPE (dw 5x5 on V) + attention residual -> xa (NHWC) --------------
__global__ void lepe_res_kernel(const float* __restrict__ x, const float* __restrict__ obuf,
                                const f16* __restrict__ v_h,
                                const float* __restrict__ w_lepe, const float* __restrict__ b_lepe,
                                float* __restrict__ xa) {
  size_t t = (size_t)blockIdx.x * 256 + threadIdx.x;
  int c = t & 255;
  size_t pix = t >> 8;
  int xx = pix & 63, y = (pix >> 6) & 63, n = (int)(pix >> 12);
  float acc = b_lepe[c];
#pragma unroll
  for (int ky = 0; ky < 5; ++ky) {
    int yy = y + ky - 2;
    if (yy < 0 || yy > 63) continue;
#pragma unroll
    for (int kx = 0; kx < 5; ++kx) {
      int x2 = xx + kx - 2;
      if (x2 < 0 || x2 > 63) continue;
      int win = (yy >> 2) * 16 + (x2 >> 2), tok = (yy & 3) * 4 + (x2 & 3);
      acc += w_lepe[c * 25 + ky * 5 + kx] *
             (float)v_h[(((size_t)(n * P2 + win)) * W2 + tok) * DIM + c];
    }
  }
  xa[t] = x[(((size_t)n * DIM + c) * 64 + y) * 64 + xx] + obuf[t] + acc;
}

// ---- 6) MLP fc1: 32 pixels x 256 -> 1024, BN, f16 -------------------------
// K-outer, two passes of 4 n-tiles; each B frag feeds 2 WMMAs.
__global__ void __launch_bounds__(256, 1)
fc1_kernel(const float* __restrict__ xa, const f16* __restrict__ w1p,
           const float* __restrict__ g1, const float* __restrict__ b1,
           f16* __restrict__ h1) {
  __shared__ __align__(32) f16 a_lds[2][16 * DIM];  // 16KB
  int mt = blockIdx.x, tid = threadIdx.x;
  const float* ab = xa + (size_t)mt * 32 * DIM;
#pragma unroll
  for (int t = 0; t < 32; ++t)
    a_lds[t >> 4][pack_idx(t & 15, tid)] = (f16)ab[(size_t)t * DIM + tid];
  __syncthreads();
  int wave = tid >> 5, lane = tid & 31, r = lane & 15, hlf = lane >> 4;
  for (int half = 0; half < 2; ++half) {
    const f16* bp[4];
#pragma unroll
    for (int j = 0; j < 4; ++j)
      bp[j] = w1p + (size_t)(wave + (half * 4 + j) * 8) * 4096 + lane * 16;
    __builtin_prefetch(bp[0], 0, 0);
    v8f acc[4][2] = {};
#pragma unroll 2
    for (int kk = 0; kk < 8; ++kk) {
      v16h a0 = frag_ld16(&a_lds[0][kk * 512] + lane * 16);
      v16h a1 = frag_ld16(&a_lds[1][kk * 512] + lane * 16);
#pragma unroll
      for (int j = 0; j < 4; ++j) {
        v16h b = frag_ld16(bp[j] + kk * 512);
        acc[j][0] = wmma16(a0, b, acc[j][0]);
        acc[j][1] = wmma16(a1, b, acc[j][1]);
      }
    }
#pragma unroll
    for (int j = 0; j < 4; ++j) {
      int n0 = (wave + (half * 4 + j) * 8) * 16;
      float gg = g1[n0 + r], be = b1[n0 + r];
#pragma unroll
      for (int mi = 0; mi < 2; ++mi) {
        union { v8f v; float f[8]; } ua; ua.v = acc[j][mi];
#pragma unroll
        for (int i2 = 0; i2 < 8; ++i2)
          h1[((size_t)mt * 32 + mi * 16 + i2 + 8 * hlf) * HID + n0 + r] =
              (f16)(ua.f[i2] * gg + be);
      }
    }
  }
}

// ---- 7) depthwise 3x3 + ReLU ---------------------------------------------
__global__ void dw3_kernel(const f16* __restrict__ h1, const float* __restrict__ wdw,
                           const float* __restrict__ bdw, f16* __restrict__ h2) {
  size_t t = (size_t)blockIdx.x * 256 + threadIdx.x;
  int c = t & 1023;
  size_t pix = t >> 10;
  int xx = pix & 63, y = (pix >> 6) & 63, n = (int)(pix >> 12);
  float acc = bdw[c];
#pragma unroll
  for (int ky = 0; ky < 3; ++ky) {
    int yy = y + ky - 1;
    if (yy < 0 || yy > 63) continue;
#pragma unroll
    for (int kx = 0; kx < 3; ++kx) {
      int x2 = xx + kx - 1;
      if (x2 < 0 || x2 > 63) continue;
      acc += wdw[c * 9 + ky * 3 + kx] *
             (float)h1[(((size_t)n * HW) + yy * 64 + x2) * HID + c];
    }
  }
  h2[t] = (f16)fmaxf(acc, 0.f);
}

// ---- 8) MLP fc2 + BN + residual, NCHW fp32 out ----------------------------
// 32 pixels per block; K=1024 in two 512-wide LDS stages; K-outer inner loop
// does 2 A loads + 2 B loads -> 4 WMMAs; acc[2][2] register-resident.
__global__ void __launch_bounds__(256, 1)
fc2_kernel(const f16* __restrict__ h2, const f16* __restrict__ w2p,
           const float* __restrict__ g2, const float* __restrict__ b2,
           const float* __restrict__ xa, float* __restrict__ out) {
  __shared__ __align__(32) f16 a_lds[2][16 * 512];  // 32KB per K-stage
  int mt = blockIdx.x, tid = threadIdx.x;
  const f16* ab = h2 + (size_t)mt * 32 * HID;
  int wave = tid >> 5, lane = tid & 31, r = lane & 15, hlf = lane >> 4;
  const f16* bp0 = w2p + (size_t)(wave * 2 + 0) * 16384 + lane * 16;
  const f16* bp1 = w2p + (size_t)(wave * 2 + 1) * 16384 + lane * 16;
  v8f acc[2][2] = {};
  for (int ks = 0; ks < 2; ++ks) {
    __syncthreads();
#pragma unroll
    for (int rep = 0; rep < 64; ++rep) {  // 32 rows x 512 cols
      int l = rep * 256 + tid;
      int row = l >> 9, col = l & 511;
      a_lds[row >> 4][pack_idx(row & 15, col)] = ab[(size_t)row * HID + ks * 512 + col];
    }
    __syncthreads();
#pragma unroll 4
    for (int kk = 0; kk < 16; ++kk) {
      v16h a0 = frag_ld16(&a_lds[0][kk * 512] + lane * 16);
      v16h a1 = frag_ld16(&a_lds[1][kk * 512] + lane * 16);
      v16h b0 = frag_ld16(bp0 + ks * 8192 + kk * 512);
      v16h b1 = frag_ld16(bp1 + ks * 8192 + kk * 512);
      acc[0][0] = wmma16(a0, b0, acc[0][0]);
      acc[0][1] = wmma16(a1, b0, acc[0][1]);
      acc[1][0] = wmma16(a0, b1, acc[1][0]);
      acc[1][1] = wmma16(a1, b1, acc[1][1]);
    }
  }
  for (int tt = 0; tt < 2; ++tt) {
    int n0 = (wave * 2 + tt) * 16, c = n0 + r;
    float gg = g2[c], be = b2[c];
#pragma unroll
    for (int mi = 0; mi < 2; ++mi) {
      union { v8f v; float f[8]; } ua; ua.v = acc[tt][mi];
#pragma unroll
      for (int i2 = 0; i2 < 8; ++i2) {
        size_t ph = (size_t)mt * 32 + mi * 16 + i2 + 8 * hlf;
        float val = ua.f[i2] * gg + be + xa[ph * DIM + c];
        int xx = ph & 63, y = (ph >> 6) & 63, n = (int)(ph >> 12);
        out[(((size_t)n * DIM + c) * 64 + y) * 64 + xx] = val;
      }
    }
  }
}

// ---- launcher -------------------------------------------------------------
extern "C" void kernel_launch(void* const* d_in, const int* in_sizes, int n_in,
                              void* d_out, int out_size, void* d_ws, size_t ws_size,
                              hipStream_t stream) {
  const float* x      = (const float*)d_in[0];
  const float* wq     = (const float*)d_in[1];
  const float* gq     = (const float*)d_in[2];
  const float* bq     = (const float*)d_in[3];
  const float* wkv    = (const float*)d_in[4];
  const float* gkv    = (const float*)d_in[5];
  const float* bkv    = (const float*)d_in[6];
  const float* w_lepe = (const float*)d_in[7];
  const float* b_lepe = (const float*)d_in[8];
  const float* w_fc1  = (const float*)d_in[9];
  const float* g_fc1  = (const float*)d_in[10];
  const float* b_fc1  = (const float*)d_in[11];
  const float* w_dw   = (const float*)d_in[12];
  const float* b_dw   = (const float*)d_in[13];
  const float* w_fc2  = (const float*)d_in[14];
  const float* g_fc2  = (const float*)d_in[15];
  const float* b_fc2  = (const float*)d_in[16];

  // workspace layout (bytes)
  constexpr size_t QP  = 0;
  constexpr size_t KP  = QP + (size_t)NB * P2 * 4096 * 2;
  constexpr size_t VP  = KP + (size_t)NB * P2 * 4096 * 2;
  constexpr size_t VH  = VP + (size_t)NB * P2 * 4096 * 2;   // linear V (LEPE)
  constexpr size_t QW  = VH + (size_t)NB * P2 * W2 * DIM * 2;
  constexpr size_t KW  = QW + (size_t)NB * P2 * QK * 4;
  constexpr size_t LG  = KW + (size_t)NB * P2 * QK * 4;
  constexpr size_t RI  = LG + (size_t)NB * P2 * P2 * 4;
  constexpr size_t OB  = RI + (size_t)NB * P2 * TOPK * 4;
  constexpr size_t XA  = OB + (size_t)NPIX * DIM * 4;
  constexpr size_t H1  = XA + (size_t)NPIX * DIM * 4;
  constexpr size_t WQP = H1 + (size_t)NPIX * HID * 2;
  constexpr size_t WKV = WQP + (size_t)QK * DIM * 2;
  constexpr size_t WF1 = WKV + (size_t)(QK + DIM) * DIM * 2;
  constexpr size_t WF2 = WF1 + (size_t)HID * DIM * 2;
  constexpr size_t KWP = WF2 + (size_t)DIM * HID * 2;
  constexpr size_t H2  = 0;  // 67.1MB alias over QP..VH (dead by then)

  char* ws = (char*)d_ws;
  f16*   qp   = (f16*)(ws + QP);
  f16*   kp   = (f16*)(ws + KP);
  f16*   vp   = (f16*)(ws + VP);
  f16*   v_h  = (f16*)(ws + VH);
  float* qwin = (float*)(ws + QW);
  float* kwin = (float*)(ws + KW);
  float* lgt  = (float*)(ws + LG);
  int*   ridx = (int*)(ws + RI);
  float* obuf = (float*)(ws + OB);
  float* xa   = (float*)(ws + XA);
  f16*   h1   = (f16*)(ws + H1);
  f16*   wqp  = (f16*)(ws + WQP);
  f16*   wkvp = (f16*)(ws + WKV);
  f16*   w1p  = (f16*)(ws + WF1);
  f16*   w2p  = (f16*)(ws + WF2);
  f16*   kwp  = (f16*)(ws + KWP);
  f16*   h2   = (f16*)(ws + H2);
  float* out  = (float*)d_out;

  dim3 blk(256);
  pack_b_kernel<<<(QK * DIM + 255) / 256, blk, 0, stream>>>(wq, wqp, DIM, (long)QK * DIM);
  pack_b_kernel<<<((QK + DIM) * DIM + 255) / 256, blk, 0, stream>>>(wkv, wkvp, DIM, (long)(QK + DIM) * DIM);
  pack_b_kernel<<<(HID * DIM + 255) / 256, blk, 0, stream>>>(w_fc1, w1p, DIM, (long)HID * DIM);
  pack_b_kernel<<<(DIM * HID + 255) / 256, blk, 0, stream>>>(w_fc2, w2p, HID, (long)DIM * HID);

  proj_kernel<<<NB * P2 / 2, blk, 0, stream>>>(x, wqp, wkvp, gq, bq, gkv, bkv,
                                               qp, kp, vp, v_h, qwin, kwin);
  pack_b_kernel<<<(NB * P2 * QK + 255) / 256, blk, 0, stream>>>(kwin, kwp, QK, (long)NB * P2 * QK);
  route_kernel<<<NB * 16, blk, 0, stream>>>(qwin, kwp, lgt);
  topk_kernel<<<(NB * P2 + 255) / 256, blk, 0, stream>>>(lgt, ridx);
  attn_kernel<<<NB * P2, blk, 0, stream>>>(qp, kp, vp, ridx, obuf);
  lepe_res_kernel<<<(size_t)NPIX * DIM / 256, blk, 0, stream>>>(x, obuf, v_h,
                                                                w_lepe, b_lepe, xa);
  fc1_kernel<<<NPIX / 32, blk, 0, stream>>>(xa, w1p, g_fc1, b_fc1, h1);
  dw3_kernel<<<(size_t)NPIX * HID / 256, blk, 0, stream>>>(h1, w_dw, b_dw, h2);
  fc2_kernel<<<NPIX / 32, blk, 0, stream>>>(h2, w2p, g_fc2, b_fc2, xa, out);
}
